// GCNPool_73770358276476
// MI455X (gfx1250) — matrix operand order repaired and checked by reading, not compiled, not run on hardware
//
#include <hip/hip_runtime.h>
#include <hip/hip_bf16.h>
#include <math.h>

typedef __attribute__((ext_vector_type(16))) _Float16 v16h;
typedef __attribute__((ext_vector_type(8)))  float    v8f;

#define DEV __device__ __forceinline__

// ---- WMMA fragment helpers (CDNA5 16x16x32 f16, wave32 layouts per ISA 7.12.2) ----
// A (16x32 f16): lane m = lane&15, hi = lane>>4; VGPR j (pair 2j,2j+1):
//   K = (j/4)*16 + hi*8 + 2*(j%4) + {0,1}
template <typename F>
DEV v16h frag_a(F f, int kc) {
  int lane = threadIdx.x & 31;
  int m = lane & 15, hi = lane >> 4;
  v16h a;
#pragma unroll
  for (int j = 0; j < 8; ++j) {
    int g = j >> 2, jj = j & 3;
    int k = kc + g * 16 + hi * 8 + 2 * jj;
    a[2 * j]     = (_Float16)f(m, k);
    a[2 * j + 1] = (_Float16)f(m, k + 1);
  }
  return a;
}
// B (32x16 f16): lane n = lane&15, hi = lane>>4; VGPR j: K = (j/4)*16 + hi*8 + 2*(j%4) + {0,1}
template <typename F>
DEV v16h frag_b(F f, int kc) {
  int lane = threadIdx.x & 31;
  int n = lane & 15, hi = lane >> 4;
  v16h b;
#pragma unroll
  for (int j = 0; j < 8; ++j) {
    int k = kc + (j >> 2) * 16 + hi * 8 + 2 * (j & 3);
    b[2 * j]     = (_Float16)f(k, n);
    b[2 * j + 1] = (_Float16)f(k + 1, n);
  }
  return b;
}

DEV v8f wmma16(v16h a, v16h b, v8f c) {
  return __builtin_amdgcn_wmma_f32_16x16x32_f16(false, a, false, b, (short)0, c,
                                                false, false);
}

// C/D (16x16 f32): lane n = lane&15; VGPR r: m = (lane>>4)*8 + r
template <typename F>
DEV void store_c(F f, v8f c) {
  int lane = threadIdx.x & 31;
  int n = lane & 15, hi = lane >> 4;
#pragma unroll
  for (int r = 0; r < 8; ++r) f(hi * 8 + r, n, c[r]);
}

DEV float sigmoidf(float x) { return 1.f / (1.f + expf(-x)); }

// ============================ Stage kernels ============================

// residual[b,o,n,l] = sum_c W1[o,c]*x[b,c,n,l] + b1[o]   (cols = n*54+l, 11178)
__global__ void k_res(const float* __restrict__ x, const float* __restrict__ W,
                      const float* __restrict__ bias, float* __restrict__ R) {
  const int COLS = 207 * 54;
  int m0 = (threadIdx.x >> 5) * 16;
  int c0 = blockIdx.x * 16;
  int b  = blockIdx.y;
  const float* xb = x + (size_t)b * 64 * COLS;
  v8f acc = {};
  for (int kc = 0; kc < 64; kc += 32) {
    v16h a  = frag_a([&](int m, int k) { return W[(m0 + m) * 64 + k]; }, kc);
    v16h bf = frag_b([&](int k, int n) {
      int col = c0 + n; if (col >= COLS) col = COLS - 1;
      return xb[(size_t)k * COLS + col];
    }, kc);
    acc = wmma16(a, bf, acc);
  }
  store_c([&](int m, int n, float v) {
    int col = c0 + n;
    if (col < COLS) R[((size_t)b * 64 + m0 + m) * COLS + col] = v + bias[m0 + m];
  }, acc);
}

// LN over time axis (54) -> xcat slab1 ; also copy residual -> slab0
__global__ void k_lnT(const float* __restrict__ R, const float* __restrict__ gT,
                      const float* __restrict__ bT, float* __restrict__ XCAT) {
  int idx = blockIdx.x * blockDim.x + threadIdx.x;
  if (idx >= 8 * 64 * 207) return;
  int b = idx / (64 * 207); int r = idx % (64 * 207);
  int o = r / 207, n = r % 207;
  const float* p = R + (((size_t)b * 64 + o) * 207 + n) * 54;
  float s = 0.f;
  for (int t = 0; t < 54; ++t) s += p[t];
  float mean = s / 54.f, s2 = 0.f;
  for (int t = 0; t < 54; ++t) { float d = p[t] - mean; s2 += d * d; }
  float rstd = rsqrtf(s2 / 54.f + 1e-5f);
  float* x0 = XCAT + (((size_t)b * 192 + o) * 207 + n) * 54;
  float* x1 = XCAT + (((size_t)b * 192 + 64 + o) * 207 + n) * 54;
  for (int t = 0; t < 54; ++t) {
    x0[t] = p[t];
    x1[t] = (p[t] - mean) * rstd * gT[t] + bT[t];
  }
}

// LN over node axis (207) -> xcat slab2
__global__ void k_lnS(const float* __restrict__ R, const float* __restrict__ gS,
                      const float* __restrict__ bS, float* __restrict__ XCAT) {
  int idx = blockIdx.x * blockDim.x + threadIdx.x;
  if (idx >= 8 * 64 * 54) return;
  int b = idx / (64 * 54); int r = idx % (64 * 54);
  int o = r / 54, l = r % 54;
  const float* p = R + (((size_t)b * 64 + o) * 207) * 54 + l;
  float s = 0.f;
  for (int n = 0; n < 207; ++n) s += p[n * 54];
  float mean = s / 207.f, s2 = 0.f;
  for (int n = 0; n < 207; ++n) { float d = p[n * 54] - mean; s2 += d * d; }
  float rstd = rsqrtf(s2 / 207.f + 1e-5f);
  float* x2 = XCAT + (((size_t)b * 192 + 128 + o) * 207) * 54 + l;
  for (int n = 0; n < 207; ++n) x2[n * 54] = (p[n * 54] - mean) * rstd * gS[n] + bS[n];
}

// 1xKT conv with dilation, im2col on the fly. out: [b][128][207][Tout]
__global__ void k_tconv(const float* __restrict__ in, int Cin, int Tin, int dil,
                        int Tout, int K, const float* __restrict__ W,
                        const float* __restrict__ bias, float* __restrict__ out) {
  int cols = 207 * Tout;
  int m0 = blockIdx.y * 64 + (threadIdx.x >> 5) * 16;
  int c0 = blockIdx.x * 16;
  int b  = blockIdx.z;
  v8f acc = {};
  for (int kc = 0; kc < K; kc += 32) {
    v16h a  = frag_a([&](int m, int k) { return W[(size_t)(m0 + m) * K + k]; }, kc);
    v16h bf = frag_b([&](int k, int n) {
      int col = c0 + n; if (col >= cols) col = cols - 1;
      int ic = k / 3, kt = k % 3;
      int nn = col / Tout, t = col % Tout;
      return in[(((size_t)b * Cin + ic) * 207 + nn) * Tin + t + dil * kt];
    }, kc);
    acc = wmma16(a, bf, acc);
  }
  store_c([&](int m, int n, float v) {
    int col = c0 + n;
    if (col < cols) {
      int nn = col / Tout, t = col % Tout;
      out[(((size_t)b * 128 + m0 + m) * 207 + nn) * Tout + t] = v + bias[m0 + m];
    }
  }, acc);
}

// GLU: out[b,c,s] = tanh(in[b,c,s]) * sigmoid(in[b,c+64,s])
__global__ void k_glu(const float* __restrict__ in, float* __restrict__ out, int S) {
  int idx = blockIdx.x * blockDim.x + threadIdx.x;
  int total = 8 * 64 * S;
  if (idx >= total) return;
  int b = idx / (64 * S); int r = idx % (64 * S);
  int c = r / S, s = r % S;
  float u = in[((size_t)b * 128 + c) * S + s];
  float v = in[((size_t)b * 128 + 64 + c) * S + s];
  out[idx] = tanhf(u) * sigmoidf(v);
}

// gated GCN inner: s[b,d,col] = sum_k sigmoid(Wq_k^T x)*(Wv_k^T x)
__global__ void k_gconv_s(const float* __restrict__ X, const float* __restrict__ wq,
                          const float* __restrict__ wv, float* __restrict__ S,
                          int C, int D) {
  const int COLS = 10350;
  int d0 = blockIdx.y * 16;
  int c0 = blockIdx.x * 16;
  int b  = blockIdx.z;
  v8f acc = {};
  for (int kb = 0; kb < 8; ++kb) {
    const float* wqk = wq + (size_t)kb * C * D;
    const float* wvk = wv + (size_t)kb * C * D;
    v8f q = {}, v = {};
    for (int kc = 0; kc < C; kc += 32) {
      v16h bf = frag_b([&](int k, int n) {
        int col = c0 + n; if (col >= COLS) col = COLS - 1;
        return X[((size_t)b * C + k) * COLS + col];
      }, kc);
      v16h aq = frag_a([&](int m, int k) { return wqk[(size_t)k * D + d0 + m]; }, kc);
      q = wmma16(aq, bf, q);
      v16h av = frag_a([&](int m, int k) { return wvk[(size_t)k * D + d0 + m]; }, kc);
      v = wmma16(av, bf, v);
    }
#pragma unroll
    for (int r = 0; r < 8; ++r) acc[r] += v[r] * sigmoidf(q[r]);
  }
  store_c([&](int m, int n, float vv) {
    int col = c0 + n;
    if (col < COLS) S[((size_t)b * D + d0 + m) * COLS + col] = vv;
  }, acc);
}

// adjacency aggregation: out[b,c,kk,l] = sum_n adj[n,kk]*s[b,c,n,l] + bias[c] (opt relu)
__global__ void k_adj(const float* __restrict__ A, const float* __restrict__ S,
                      const float* __restrict__ bias, float* __restrict__ out,
                      int D, int outChans, int chanOff, int relu) {
  int r0 = blockIdx.y * 16;      // output node kk
  int c0 = blockIdx.x * 16;      // (c,l) combined col, D*50 total
  int b  = blockIdx.z;
  v8f acc = {};
  for (int kc = 0; kc < 207; kc += 32) {
    v16h af = frag_a([&](int m, int k) {
      int kk = r0 + m; if (kk > 206) kk = 206;
      return (k < 207) ? A[(size_t)k * 207 + kk] : 0.f;
    }, kc);
    v16h bf = frag_b([&](int k, int n) {
      int col = c0 + n;
      int c = col / 50, l = col % 50;
      return (k < 207) ? S[(((size_t)b * D + c) * 207 + k) * 50 + l] : 0.f;
    }, kc);
    acc = wmma16(af, bf, acc);
  }
  store_c([&](int m, int n, float v) {
    int kk = r0 + m;
    if (kk < 207) {
      int col = c0 + n;
      int c = col / 50, l = col % 50;
      float rr = v + bias[c];
      if (relu) rr = fmaxf(rr, 0.f);
      out[(((size_t)b * outChans + chanOff + c) * 207 + kk) * 50 + l] = rr;
    }
  }, acc);
}

// patch gather + positional embedding (faithful flat-index reshape)
__global__ void k_xe(const float* __restrict__ G, const float* __restrict__ emb,
                     float* __restrict__ XE) {
  int idx = blockIdx.x * blockDim.x + threadIdx.x;
  const int PERB = 207 * 12 * 512;
  if (idx >= 8 * PERB) return;
  int b = idx / PERB; int f = idx % PERB;
  int pe = f % (12 * 512);
  int p2 = pe / 512, e2 = pe % 512;
  int c = f / 19872; int rem = f % 19872;     // 19872 = 207*12*8
  int n = rem / 96;  int r = rem % 96;
  int p = r / 8, pl = r % 8;
  int t = p * 4 + pl;
  float val = (t < 48) ? G[(((size_t)b * 64 + c) * 207 + n) * 48 + t] : 0.f;
  XE[idx] = val + emb[p2 * 512 + e2];
}

// per-head QKV projection (block-diagonal over heads; shared A-fragments)
__global__ void k_qkv(const float* __restrict__ XE,
                      const float* __restrict__ Wq, const float* __restrict__ bq,
                      const float* __restrict__ Wk, const float* __restrict__ bk,
                      const float* __restrict__ Wv, const float* __restrict__ bv,
                      float* __restrict__ QS, float* __restrict__ KS,
                      float* __restrict__ VS) {
  int r0 = blockIdx.y * 16;
  int nt = blockIdx.x;
  int h = nt >> 2, n0 = (nt & 3) * 16;
  v8f aq = {}, ak = {}, av = {};
  for (int kc = 0; kc < 64; kc += 32) {
    v16h af = frag_a([&](int m, int k) {
      return XE[(size_t)(r0 + m) * 512 + h * 64 + k];
    }, kc);
    v16h bfq = frag_b([&](int k, int n) { return Wq[k * 64 + n0 + n]; }, kc);
    aq = wmma16(af, bfq, aq);
    v16h bfk = frag_b([&](int k, int n) { return Wk[k * 64 + n0 + n]; }, kc);
    ak = wmma16(af, bfk, ak);
    v16h bfv = frag_b([&](int k, int n) { return Wv[k * 64 + n0 + n]; }, kc);
    av = wmma16(af, bfv, av);
  }
  store_c([&](int m, int n, float v) {
    QS[(size_t)(r0 + m) * 512 + h * 64 + n0 + n] = v + bq[n0 + n]; }, aq);
  store_c([&](int m, int n, float v) {
    KS[(size_t)(r0 + m) * 512 + h * 64 + n0 + n] = v + bk[n0 + n]; }, ak);
  store_c([&](int m, int n, float v) {
    VS[(size_t)(r0 + m) * 512 + h * 64 + n0 + n] = v + bv[n0 + n]; }, av);
}

// scores + softmax over QUERY axis (faithful to source)
__global__ void k_scores(const float* __restrict__ QS, const float* __restrict__ KS,
                         float* __restrict__ ATT) {
  int idx = blockIdx.x * blockDim.x + threadIdx.x;
  if (idx >= 8 * 207 * 8 * 12) return;
  int kk = idx % 12; int bnh = idx / 12;
  int h = bnh % 8;  int bn = bnh / 8;
  const float* kr = KS + ((size_t)bn * 12 + kk) * 512 + h * 64;
  float sc[12];
  float mx = -1e30f;
  for (int q = 0; q < 12; ++q) {
    const float* qr = QS + ((size_t)bn * 12 + q) * 512 + h * 64;
    float s = 0.f;
    for (int d = 0; d < 64; ++d) s += qr[d] * kr[d];
    s *= 0.044194173824159216f;  // 1/sqrt(512)
    sc[q] = s; mx = fmaxf(mx, s);
  }
  float sum = 0.f;
  for (int q = 0; q < 12; ++q) { sc[q] = expf(sc[q] - mx); sum += sc[q]; }
  float inv = 1.f / sum;
  for (int q = 0; q < 12; ++q) ATT[(size_t)bnh * 144 + q * 12 + kk] = sc[q] * inv;
}

__global__ void k_ao(const float* __restrict__ ATT, const float* __restrict__ VS,
                     float* __restrict__ AO) {
  int idx = blockIdx.x * blockDim.x + threadIdx.x;
  if (idx >= 19872 * 512) return;
  int col = idx % 512; int row = idx / 512;
  int q = row % 12; int bn = row / 12;
  int h = col >> 6;
  const float* att = ATT + ((size_t)bn * 8 + h) * 144 + q * 12;
  float s = 0.f;
  for (int k = 0; k < 12; ++k) s += att[k] * VS[((size_t)bn * 12 + k) * 512 + col];
  AO[idx] = s;
}

// generic row-major GEMM: C[R x Nc] = A[R x K] @ B[K x Nc] + bias (opt relu)
__global__ void k_gemm(const float* __restrict__ A, const float* __restrict__ B,
                       const float* __restrict__ bias, float* __restrict__ C,
                       int K, int Nc, int relu) {
  int r0 = blockIdx.y * 16;
  int n0 = blockIdx.x * 16;
  v8f acc = {};
  for (int kc = 0; kc < K; kc += 32) {
    v16h af = frag_a([&](int m, int k) { return A[(size_t)(r0 + m) * K + k]; }, kc);
    v16h bf = frag_b([&](int k, int n) { return B[(size_t)k * Nc + n0 + n]; }, kc);
    acc = wmma16(af, bf, acc);
  }
  store_c([&](int m, int n, float v) {
    v += bias[n0 + n];
    if (relu) v = fmaxf(v, 0.f);
    C[(size_t)(r0 + m) * Nc + n0 + n] = v;
  }, acc);
}

// LayerNorm over E=512 of (a+b), one block per row
__global__ void k_ln(const float* __restrict__ A, const float* __restrict__ Bv,
                     const float* __restrict__ gamma, const float* __restrict__ beta,
                     float* __restrict__ out) {
  __shared__ float red[256];
  int row = blockIdx.x, t = threadIdx.x;
  const float* pa = A + (size_t)row * 512;
  const float* pb = Bv + (size_t)row * 512;
  float s = 0.f;
  for (int e = t; e < 512; e += 256) s += pa[e] + pb[e];
  red[t] = s; __syncthreads();
  for (int o = 128; o > 0; o >>= 1) { if (t < o) red[t] += red[t + o]; __syncthreads(); }
  float mean = red[0] * (1.f / 512.f);
  __syncthreads();
  float s2 = 0.f;
  for (int e = t; e < 512; e += 256) { float d = pa[e] + pb[e] - mean; s2 += d * d; }
  red[t] = s2; __syncthreads();
  for (int o = 128; o > 0; o >>= 1) { if (t < o) red[t] += red[t + o]; __syncthreads(); }
  float rstd = rsqrtf(red[0] * (1.f / 512.f) + 1e-5f);
  for (int e = t; e < 512; e += 256)
    out[(size_t)row * 512 + e] = (pa[e] + pb[e] - mean) * rstd * gamma[e] + beta[e];
}

__global__ void k_add3(const float* __restrict__ a, const float* __restrict__ b,
                       const float* __restrict__ c, float* __restrict__ o, int n) {
  int idx = blockIdx.x * blockDim.x + threadIdx.x;
  if (idx < n) o[idx] = a[idx] + b[idx] + c[idx];
}

// per-channel BN stats of y = o + residual[...,-48:] over (b,n,t)
__global__ void k_bnstat(const float* __restrict__ O, const float* __restrict__ R,
                         float* __restrict__ stat) {
  __shared__ float rs[256], rs2[256];
  int c = blockIdx.x, t = threadIdx.x;
  float s = 0.f, s2 = 0.f;
  for (int i = t; i < 8 * 207 * 48; i += 256) {
    int b = i / (207 * 48); int r = i % (207 * 48);
    int n = r / 48, tt = r % 48;
    float y = O[(((size_t)b * 207 + n) * 64 + c) * 48 + tt]
            + R[(((size_t)b * 64 + c) * 207 + n) * 54 + 6 + tt];
    s += y; s2 += y * y;
  }
  rs[t] = s; rs2[t] = s2; __syncthreads();
  for (int o = 128; o > 0; o >>= 1) {
    if (t < o) { rs[t] += rs[t + o]; rs2[t] += rs2[t + o]; }
    __syncthreads();
  }
  if (t == 0) {
    float mean = rs[0] / 79488.f;
    stat[c] = mean;
    stat[64 + c] = rs2[0] / 79488.f - mean * mean;
  }
}

__global__ void k_bnout(const float* __restrict__ O, const float* __restrict__ R,
                        const float* __restrict__ stat, const float* __restrict__ g,
                        const float* __restrict__ be, float* __restrict__ out) {
  int idx = blockIdx.x * blockDim.x + threadIdx.x;
  if (idx >= 8 * 64 * 207 * 48) return;
  int b = idx / (64 * 207 * 48); int r = idx % (64 * 207 * 48);
  int c = r / (207 * 48); int rr = r % (207 * 48);
  int n = rr / 48, t = rr % 48;
  float y = O[(((size_t)b * 207 + n) * 64 + c) * 48 + t]
          + R[(((size_t)b * 64 + c) * 207 + n) * 54 + 6 + t];
  out[idx] = (y - stat[c]) * rsqrtf(stat[64 + c] + 1e-5f) * g[c] + be[c];
}

// ============================ Launch ============================
extern "C" void kernel_launch(void* const* d_in, const int* in_sizes, int n_in,
                              void* d_out, int out_size, void* d_ws, size_t ws_size,
                              hipStream_t stream) {
  (void)in_sizes; (void)n_in; (void)out_size; (void)ws_size;
  const float* x     = (const float*)d_in[0];
  const float* adj   = (const float*)d_in[1];
  const float* W1    = (const float*)d_in[2];
  const float* b1    = (const float*)d_in[3];
  const float* gT    = (const float*)d_in[4];
  const float* bT    = (const float*)d_in[5];
  const float* gS    = (const float*)d_in[6];
  const float* bS    = (const float*)d_in[7];
  const float* Wt    = (const float*)d_in[8];
  const float* bt    = (const float*)d_in[9];
  const float* wq1   = (const float*)d_in[10];
  const float* wv1   = (const float*)d_in[11];
  const float* bias1 = (const float*)d_in[12];
  const float* wq2   = (const float*)d_in[13];
  const float* wv2   = (const float*)d_in[14];
  const float* bias2 = (const float*)d_in[15];
  const float* Wm    = (const float*)d_in[16];
  const float* bm    = (const float*)d_in[17];
  const float* embT  = (const float*)d_in[18];
  const float* Wq    = (const float*)d_in[19];
  const float* bq    = (const float*)d_in[20];
  const float* Wk    = (const float*)d_in[21];
  const float* bk    = (const float*)d_in[22];
  const float* Wv    = (const float*)d_in[23];
  const float* bv    = (const float*)d_in[24];
  const float* Wfc   = (const float*)d_in[25];
  const float* bfc   = (const float*)d_in[26];
  const float* g1    = (const float*)d_in[27];
  const float* b1n   = (const float*)d_in[28];
  const float* g2    = (const float*)d_in[29];
  const float* b2n   = (const float*)d_in[30];
  const float* Wff1  = (const float*)d_in[31];
  const float* bff1  = (const float*)d_in[32];
  const float* Wff2  = (const float*)d_in[33];
  const float* bff2  = (const float*)d_in[34];
  const float* Wrp   = (const float*)d_in[35];
  const float* brp   = (const float*)d_in[36];
  const float* gbn   = (const float*)d_in[37];
  const float* bbn   = (const float*)d_in[38];

  float* ws   = (float*)d_ws;
  float* R    = ws;                     // [8,64,207,54]
  float* XCAT = R    + 5723136;         // [8,192,207,54] ; later aliased as XE
  float* HPRE = XCAT + 17169408;        // [8,128,207,50]
  float* H    = HPRE + 10598400;        // [8,64,207,50]
  float* SBUF = H    + 5299200;         // [8,128,207,50] (max D)
  float* G1   = SBUF + 10598400;        // [8,64,207,50] ; later aliased as O
  float* O3   = G1   + 5299200;         // [8,384,207,50]
  float* GPRE = O3   + 31795200;        // [8,128,207,48] ; later aliased as S3
  float* G    = GPRE + 10174464;        // [8,64,207,48]
  float* QS   = G    + 5087232;         // [19872,512] ; later aliased as AO2
  float* KS   = QS   + 10174464;        // [19872,512] ; later aliased as M
  float* VS   = KS   + 10174464;        // [19872,512] ; later aliased as FF2
  float* ATT  = VS   + 10174464;        // [13248,144]
  float* AO   = ATT  + 1907712;         // [19872,512] ; later aliased as U
  float* FF1  = AO   + 10174464;        // [19872,256]
  float* STAT = FF1  + 5087232;         // [128]
  float* XE = XCAT; float* Ob = G1; float* S3 = GPRE;
  float* AO2 = QS;  float* Mb = KS; float* FF2 = VS; float* Ub = AO;

  k_res<<<dim3(699, 8), 128, 0, stream>>>(x, W1, b1, R);
  k_lnT<<<(8 * 64 * 207 + 255) / 256, 256, 0, stream>>>(R, gT, bT, XCAT);
  k_lnS<<<(8 * 64 * 54 + 255) / 256, 256, 0, stream>>>(R, gS, bS, XCAT);
  k_tconv<<<dim3(647, 2, 8), 128, 0, stream>>>(XCAT, 192, 54, 2, 50, 576, Wt, bt, HPRE);
  k_glu<<<(8 * 64 * 10350 + 255) / 256, 256, 0, stream>>>(HPRE, H, 10350);

  for (int i = 0; i < 3; ++i) {
    const float* ai = adj + (size_t)i * 207 * 207;
    k_gconv_s<<<dim3(647, 4, 8), 32, 0, stream>>>(
        H, wq1 + (size_t)i * 8 * 64 * 64, wv1 + (size_t)i * 8 * 64 * 64, SBUF, 64, 64);
    k_adj<<<dim3(200, 13, 8), 32, 0, stream>>>(ai, SBUF, bias1 + i * 64, G1, 64, 64, 0, 1);
    k_gconv_s<<<dim3(647, 8, 8), 32, 0, stream>>>(
        G1, wq2 + (size_t)i * 8 * 64 * 128, wv2 + (size_t)i * 8 * 64 * 128, SBUF, 64, 128);
    k_adj<<<dim3(400, 13, 8), 32, 0, stream>>>(ai, SBUF, bias2 + i * 128, O3, 128, 384,
                                               i * 128, 0);
  }

  k_tconv<<<dim3(621, 2, 8), 128, 0, stream>>>(O3, 384, 50, 1, 48, 1152, Wm, bm, GPRE);
  k_glu<<<(8 * 64 * 9936 + 255) / 256, 256, 0, stream>>>(GPRE, G, 9936);
  k_xe<<<(10174464 + 255) / 256, 256, 0, stream>>>(G, embT, XE);
  k_qkv<<<dim3(32, 1242), 32, 0, stream>>>(XE, Wq, bq, Wk, bk, Wv, bv, QS, KS, VS);
  k_scores<<<(158976 + 127) / 128, 128, 0, stream>>>(QS, KS, ATT);
  k_ao<<<(10174464 + 255) / 256, 256, 0, stream>>>(ATT, VS, AO);
  k_gemm<<<dim3(32, 1242), 32, 0, stream>>>(AO, Wfc, bfc, AO2, 512, 512, 0);
  k_ln<<<19872, 256, 0, stream>>>(AO2, XE, g1, b1n, Mb);
  k_gemm<<<dim3(16, 1242), 32, 0, stream>>>(Mb, Wff1, bff1, FF1, 512, 256, 1);
  k_gemm<<<dim3(32, 1242), 32, 0, stream>>>(FF1, Wff2, bff2, FF2, 256, 512, 0);
  k_ln<<<19872, 256, 0, stream>>>(Mb, FF2, g2, b2n, Ub);
  k_add3<<<(10174464 + 255) / 256, 256, 0, stream>>>(Ub, XE, Mb, S3, 10174464);
  k_gemm<<<dim3(3, 6624), 32, 0, stream>>>(S3, Wrp, brp, Ob, 96, 48, 0);
  k_bnstat<<<64, 256, 0, stream>>>(Ob, R, STAT);
  k_bnout<<<(4073472 + 255) / 256, 256, 0, stream>>>(Ob, R, STAT, gbn, bbn,
                                                     (float*)d_out);
}
